// Linear3Bit_29068338659340
// MI455X (gfx1250) — matrix operand order repaired and checked by reading, not compiled, not run on hardware
//
#include <hip/hip_runtime.h>

typedef __attribute__((ext_vector_type(16))) _Float16 v16h;
typedef __attribute__((ext_vector_type(8)))  _Float16 v8h;
typedef __attribute__((ext_vector_type(8)))  float    v8f;
typedef __attribute__((ext_vector_type(4)))  float    v4f;

#define IN_F   4096
#define OUT_F  4096
#define BATCH  32
#define KTILE  256
#define NTILE  32
#define THREADS 128
#define GROUPS_PER_ROW (IN_F / 8)   // 512 three-byte groups per output row

// Block: 128 threads = 4 wave32. Waves tile the 32x32 output:
//   m_tile = wave & 1   (rows  m_tile*16 .. +15)
//   n_tile = wave >> 1  (cols  n0 + n_tile*16 .. +15)
// Grid: OUT_F / NTILE = 128 blocks.
__global__ __launch_bounds__(THREADS) void l3b_wmma_kernel(
    const float*    __restrict__ x,      // [32, 4096] f32
    const int*      __restrict__ wq3,    // [N_GROUPS, 3] int32 (byte values)
    const _Float16* __restrict__ wnorm,  // [N_GROUPS] f16 scales
    const float*    __restrict__ bias,   // [4096] f32
    float*          __restrict__ out)    // [32, 4096] f32
{
    __shared__ _Float16 x_lds[BATCH * KTILE];   // 16 KB, [row][k] k-contiguous
    __shared__ _Float16 w_lds[NTILE * KTILE];   // 16 KB, [n_local][k] k-contiguous

    const int tid    = threadIdx.x;
    const int lane   = tid & 31;
    const int wave   = tid >> 5;
    const int laneHi = lane >> 4;   // 0: lanes 0-15, 1: lanes 16-31
    const int lane16 = lane & 15;

    const int n0     = blockIdx.x * NTILE;
    const int m_tile = wave & 1;
    const int n_tile = wave >> 1;

    v8f acc = {};   // 16x16 f32 accumulator (8 VGPRs, wave32 layout)

    for (int kb = 0; kb < IN_F; kb += KTILE) {
        // ---- stage X tile (32 x 256) f32 -> f16, coalesced float4 loads ----
        // 32*256 = 8192 elems = 2048 float4; 128 threads -> 16 each
        #pragma unroll
        for (int it = 0; it < 16; ++it) {
            int idx4 = it * THREADS + tid;        // 0..2047
            int row  = idx4 >> 6;                 // 64 float4 per row
            int k4   = (idx4 & 63) << 2;
            v4f xv = *(const v4f*)(x + row * IN_F + kb + k4);
            _Float16* dst = &x_lds[row * KTILE + k4];
            dst[0] = (_Float16)xv.x; dst[1] = (_Float16)xv.y;
            dst[2] = (_Float16)xv.z; dst[3] = (_Float16)xv.w;
        }

        // ---- dequant weight tile (32 rows x 32 k-groups of 8) ----
        // 1024 groups; 128 threads -> 8 each; consecutive tid -> consecutive g
        #pragma unroll
        for (int it = 0; it < 8; ++it) {
            int gt      = it * THREADS + tid;     // 0..1023
            int n_local = gt >> 5;                // 0..31
            int kg      = gt & 31;                // k-group within tile
            int g = (n0 + n_local) * GROUPS_PER_ROW + (kb >> 3) + kg;
            const int* q = wq3 + g * 3;
            int word = (q[0] & 255) | ((q[1] & 255) << 8) | ((q[2] & 255) << 16);
            float n     = (float)wnorm[g];
            float scale = n * (2.0f / 7.0f);
            v8h wv;
            #pragma unroll
            for (int i = 0; i < 8; ++i)
                wv[i] = (_Float16)((float)((word >> (3 * i)) & 7) * scale - n);
            *(v8h*)&w_lds[n_local * KTILE + kg * 8] = wv;   // 16B aligned store
        }
        __syncthreads();

        // hint the next weight tile into cache while we do the math
        if (kb + KTILE < IN_F)
            __builtin_prefetch(
                wq3 + ((size_t)n0 * GROUPS_PER_ROW + ((kb + KTILE) >> 3)) * 3, 0, 1);

        // ---- WMMA over the K tile: 8 x v_wmma_f32_16x16x32_f16 ----
        #pragma unroll
        for (int kk = 0; kk < KTILE; kk += 32) {
            // A fragment (16x32 f16): lane<16 holds M=lane16, K = kk+0..15
            //                         lane>=16 same M,      K = kk+16..31
            const _Float16* arow = &x_lds[(m_tile * 16 + lane16) * KTILE];
            union { struct { v8h lo, hi; } s; v16h v; } ua;
            ua.s.lo = *(const v8h*)(arow + kk + laneHi * 16);       // ds_load_b128
            ua.s.hi = *(const v8h*)(arow + kk + 8 + laneHi * 16);   // ds_load_b128
            // B fragment (32x16 f16): lanes 0-15 hold N=lane16, K = kk..kk+15,
            //                         lanes 16-31 hold same N,  K = kk+16..kk+31
            const _Float16* brow = &w_lds[(n_tile * 16 + lane16) * KTILE];
            v16h bfrag = *(const v16h*)(brow + kk + laneHi * 16);   // 32B contig

            acc = __builtin_amdgcn_wmma_f32_16x16x32_f16(
                /*neg_a=*/false, ua.v, /*neg_b=*/false, bfrag,
                /*c_mod=*/(short)0, acc, /*reuse_a=*/false, /*reuse_b=*/false);
        }
        __syncthreads();
    }

    // ---- epilogue: C/D layout is VGPR r -> M=r (lanes 0-15), M=r+8 (16-31) ----
    const int col = n0 + n_tile * 16 + lane16;
    const float b = bias[col];
    #pragma unroll
    for (int r = 0; r < 8; ++r) {
        int row = m_tile * 16 + r + laneHi * 8;
        out[row * OUT_F + col] = acc[r] + b;
    }
}

extern "C" void kernel_launch(void* const* d_in, const int* in_sizes, int n_in,
                              void* d_out, int out_size, void* d_ws, size_t ws_size,
                              hipStream_t stream) {
    const float*    x     = (const float*)d_in[0];
    const int*      wq3   = (const int*)d_in[1];
    const _Float16* wnorm = (const _Float16*)d_in[2];
    const float*    bias  = (const float*)d_in[3];
    float*          out   = (float*)d_out;

    dim3 grid(OUT_F / NTILE);   // 128 blocks
    dim3 block(THREADS);        // 128 threads = 4 wave32
    hipLaunchKernelGGL(l3b_wmma_kernel, grid, block, 0, stream,
                       x, wq3, wnorm, bias, out);
}
